// GAT_3899830304826
// MI455X (gfx1250) — compile-verified, matrix operand order checked by reading, not run on hardware
//
#include <hip/hip_runtime.h>

// Problem constants (match reference)
#define BB 2
#define NN 2048
#define FF 256
#define OUTC 256
#define HH 4
#define NEG_SLOPE 0.2f

typedef __attribute__((ext_vector_type(16))) __bf16 v16bf;
typedef __attribute__((ext_vector_type(8)))  float  v8f;

struct __align__(16) U4 { unsigned int x, y, z, w; };
struct __align__(8)  F2 { float x, y; };

union V16 { U4 q[2]; v16bf b; };

__device__ __forceinline__ unsigned short f2bf(float x) {
  unsigned int u = __float_as_uint(x);
  unsigned int r = ((u >> 16) & 1u) + 0x7fffu;   // round-to-nearest-even
  return (unsigned short)((u + r) >> 16);
}
__device__ __forceinline__ float bf2f(unsigned short h) {
  return __uint_as_float(((unsigned int)h) << 16);
}

// ---------------------------------------------------------------------------
// Tensor Data Mover support (gfx1250). Guarded: falls back to plain loads.
// ---------------------------------------------------------------------------
#if defined(__AMDGCN__) && __has_builtin(__builtin_amdgcn_tensor_load_to_lds)
#define HAVE_TDM 1

typedef unsigned int tdm_v4u __attribute__((ext_vector_type(4)));
typedef int          tdm_v8i __attribute__((ext_vector_type(8)));
typedef int          tdm_v4i __attribute__((ext_vector_type(4)));

// 2D f32 tile load: tile_rows x tile_cols from a row-major tensor with
// row stride `stride_elems`, into LDS at `lds_dst` (rows stored contiguously).
// D# packing per CDNA5 ISA (groups 0/1; groups 2/3 unused).
__device__ __forceinline__ void tdm_load_tile_f32(const float* gsrc, float* lds_dst,
                                                  int tile_rows, int tile_cols,
                                                  int stride_elems, int tensor_rows) {
  unsigned long long ga = (unsigned long long)(size_t)gsrc;
  unsigned int lds_off =
      (unsigned int)(unsigned long long)(__attribute__((address_space(3))) float*)lds_dst;

  tdm_v4u g0;
  g0[0] = 1u;                                          // count=1, user descriptor
  g0[1] = lds_off;                                     // lds_addr (bytes)
  g0[2] = (unsigned int)(ga & 0xFFFFFFFFu);            // global_addr[31:0]
  g0[3] = (unsigned int)((ga >> 32) & 0x01FFFFFFu)     // global_addr[56:32]
          | (2u << 30);                                // type = 2 ("image")

  unsigned int d0 = (unsigned int)stride_elems;        // tensor_dim0
  unsigned int d1 = (unsigned int)tensor_rows;         // tensor_dim1
  unsigned int s0 = (unsigned int)stride_elems;        // tensor_dim0_stride
  tdm_v8i g1;
  g1[0] = (int)(2u << 16);                             // data_size=2 (4B), no multicast
  g1[1] = (int)((d0 & 0xFFFFu) << 16);                 // [47:32]=0, [63:48]=dim0 lo
  g1[2] = (int)(((d0 >> 16) & 0xFFFFu) | ((d1 & 0xFFFFu) << 16));
  g1[3] = (int)(((d1 >> 16) & 0xFFFFu) | ((unsigned)tile_cols << 16)); // tile_dim0
  g1[4] = (int)((unsigned)tile_rows);                  // tile_dim1; tile_dim2=0
  g1[5] = (int)(s0);                                   // stride0[31:0]
  g1[6] = 0;                                           // stride0 hi / stride1 lo
  g1[7] = 0;

  tdm_v4i z4 = {0, 0, 0, 0};
#if __clang_major__ >= 23
  tdm_v8i z8 = {0, 0, 0, 0, 0, 0, 0, 0};
  __builtin_amdgcn_tensor_load_to_lds(g0, g1, z4, z4, z8, 0);
#else
  __builtin_amdgcn_tensor_load_to_lds(g0, g1, z4, z4, 0);
#endif
}
#endif

// ---------------------------------------------------------------------------
// Kernel 1a: inputs f32 -> bf16 (same layout [B][N][F])
// ---------------------------------------------------------------------------
__global__ __launch_bounds__(256) void k_cvt_inputs(const float* __restrict__ x,
                                                    unsigned short* __restrict__ y,
                                                    int n) {
  int i = blockIdx.x * 256 + threadIdx.x;
  if (i < n) y[i] = f2bf(x[i]);
}

// ---------------------------------------------------------------------------
// Kernel 1b: W [H][F][OUT] f32 -> W_T [H][OUT][F] bf16
// ---------------------------------------------------------------------------
__global__ __launch_bounds__(256) void k_tr_W(const float* __restrict__ W,
                                              unsigned short* __restrict__ wT) {
  int i = blockIdx.x * 256 + threadIdx.x;     // i < H*F*OUT
  int h = i / (FF * OUTC);
  int rem = i % (FF * OUTC);
  int f = rem / OUTC;
  int o = rem % OUTC;
  wT[((size_t)h * OUTC + o) * FF + f] = f2bf(W[i]);
}

// ---------------------------------------------------------------------------
// Kernel 2: fts = inputs @ W per head, via WMMA bf16, stored TRANSPOSED:
//           ftsT[h][b][o][m]  (bf16, m contiguous)
// ---------------------------------------------------------------------------
__global__ __launch_bounds__(256) void k_fts(const unsigned short* __restrict__ in_bf,
                                             const unsigned short* __restrict__ wT,
                                             unsigned short* __restrict__ ftsT) {
  const int nt = blockIdx.x % (NN / 16);
  const int b  = (blockIdx.x / (NN / 16)) % BB;
  const int h  = blockIdx.x / ((NN / 16) * BB);
  const int tid  = threadIdx.x;
  const int wave = tid >> 5, lane = tid & 31, lo = lane & 15, hi = lane >> 4;
  const int rowbase = nt * 16;

  v8f acc[2] = {};
  const unsigned short* arow = in_bf + ((size_t)b * NN + rowbase + lo) * FF;

  for (int f = 0; f < FF; f += 32) {
    V16 A;
    A.q[0] = *(const U4*)(arow + f + hi * 8);
    A.q[1] = *(const U4*)(arow + f + hi * 8 + 16);
#pragma unroll
    for (int t = 0; t < 2; ++t) {
      const int col = wave * 32 + t * 16 + lo;
      const unsigned short* bp = wT + ((size_t)h * OUTC + col) * FF + f + hi * 8;
      V16 Bv;
      Bv.q[0] = *(const U4*)(bp);
      Bv.q[1] = *(const U4*)(bp + 16);
      acc[t] = __builtin_amdgcn_wmma_f32_16x16x32_bf16(false, A.b, false, Bv.b,
                                                       (short)0, acc[t], false, false);
    }
  }

#pragma unroll
  for (int t = 0; t < 2; ++t) {
    const int col = wave * 32 + t * 16 + lo;
    U4 d;
    d.x = (unsigned)f2bf(acc[t][0]) | ((unsigned)f2bf(acc[t][1]) << 16);
    d.y = (unsigned)f2bf(acc[t][2]) | ((unsigned)f2bf(acc[t][3]) << 16);
    d.z = (unsigned)f2bf(acc[t][4]) | ((unsigned)f2bf(acc[t][5]) << 16);
    d.w = (unsigned)f2bf(acc[t][6]) | ((unsigned)f2bf(acc[t][7]) << 16);
    *(U4*)(ftsT + (((size_t)h * BB + b) * OUTC + col) * NN + rowbase + hi * 8) = d;
  }
}

// ---------------------------------------------------------------------------
// Kernel 3: f1[h,b,n] = fts . w1 + b1 ; f2 likewise. Reads ftsT columns.
// ---------------------------------------------------------------------------
__global__ __launch_bounds__(256) void k_f1f2(const unsigned short* __restrict__ ftsT,
                                              const float* __restrict__ w1,
                                              const float* __restrict__ b1,
                                              const float* __restrict__ w2,
                                              const float* __restrict__ b2,
                                              float* __restrict__ f1,
                                              float* __restrict__ f2) {
  int idx = blockIdx.x * 256 + threadIdx.x;      // idx < H*B*N
  int h = idx / (BB * NN);
  int rem = idx % (BB * NN);
  int b = rem / NN;
  int n = rem % NN;
  const unsigned short* base = ftsT + ((size_t)h * BB + b) * OUTC * NN + n;
  float a1 = 0.f, a2 = 0.f;
  for (int o = 0; o < OUTC; ++o) {
    float v = bf2f(base[(size_t)o * NN]);
    a1 += v * w1[h * OUTC + o];
    a2 += v * w2[h * OUTC + o];
  }
  f1[idx] = a1 + b1[h];
  f2[idx] = a2 + b2[h];
}

// ---------------------------------------------------------------------------
// Kernel 4: per-row softmax statistics (shared adj_seq + 4 head logits).
// ---------------------------------------------------------------------------
__global__ __launch_bounds__(256) void k_stats(const float* __restrict__ adj,
                                               const float* __restrict__ adj_seq,
                                               const float* __restrict__ f1,
                                               const float* __restrict__ f2,
                                               float* __restrict__ rmax,
                                               float* __restrict__ rsum,
                                               float* __restrict__ rmax2,
                                               float* __restrict__ rsum2) {
  const int b = blockIdx.x / NN;
  const int n = blockIdx.x % NN;
  const int tid = threadIdx.x;

  __shared__ float s_as[NN];
  __shared__ float s_adj[NN];
  __shared__ float red[256];

  auto redmax = [&](float v) -> float {
    red[tid] = v; __syncthreads();
    for (int s = 128; s > 0; s >>= 1) {
      if (tid < s) red[tid] = fmaxf(red[tid], red[tid + s]);
      __syncthreads();
    }
    float r = red[0]; __syncthreads(); return r;
  };
  auto redsum = [&](float v) -> float {
    red[tid] = v; __syncthreads();
    for (int s = 128; s > 0; s >>= 1) {
      if (tid < s) red[tid] = red[tid] + red[tid + s];
      __syncthreads();
    }
    float r = red[0]; __syncthreads(); return r;
  };

  float f1v[HH];
#pragma unroll
  for (int h = 0; h < HH; ++h) f1v[h] = f1[((size_t)h * BB + b) * NN + n];

  const size_t rowoff = ((size_t)b * NN + n) * NN;
  float f2c[8][HH];
  float lmax2 = -3.4e38f;
  float lmaxh[HH];
#pragma unroll
  for (int h = 0; h < HH; ++h) lmaxh[h] = -3.4e38f;

  for (int i = 0; i < 8; ++i) {
    int m = tid + i * 256;
    float as = adj_seq[rowoff + m];
    float av = adj[rowoff + m];
    s_as[m] = as; s_adj[m] = av;
    lmax2 = fmaxf(lmax2, as);
#pragma unroll
    for (int h = 0; h < HH; ++h) {
      float fv = f2[((size_t)h * BB + b) * NN + m];
      f2c[i][h] = fv;
      float x = f1v[h] + fv;
      float l = (x > 0.f) ? x : NEG_SLOPE * x;
      l += -1e9f * (1.f - av);
      lmaxh[h] = fmaxf(lmaxh[h], l);
    }
  }

  float gmax2 = redmax(lmax2);
  float gmaxh[HH];
#pragma unroll
  for (int h = 0; h < HH; ++h) gmaxh[h] = redmax(lmaxh[h]);

  float ls2 = 0.f, lsh[HH] = {0.f, 0.f, 0.f, 0.f};
  for (int i = 0; i < 8; ++i) {
    int m = tid + i * 256;
    float as = s_as[m];
    float av = s_adj[m];
    ls2 += __expf(as - gmax2);
#pragma unroll
    for (int h = 0; h < HH; ++h) {
      float x = f1v[h] + f2c[i][h];
      float l = (x > 0.f) ? x : NEG_SLOPE * x;
      l += -1e9f * (1.f - av);
      lsh[h] += __expf(l - gmaxh[h]);
    }
  }
  float gs2 = redsum(ls2);
  float gsh[HH];
#pragma unroll
  for (int h = 0; h < HH; ++h) gsh[h] = redsum(lsh[h]);

  if (tid == 0) {
    rmax2[(size_t)b * NN + n] = gmax2;
    rsum2[(size_t)b * NN + n] = gs2;
#pragma unroll
    for (int h = 0; h < HH; ++h) {
      rmax[((size_t)h * BB + b) * NN + n] = gmaxh[h];
      rsum[((size_t)h * BB + b) * NN + n] = gsh[h];
    }
  }
}

// ---------------------------------------------------------------------------
// Kernel 5: fused attention aggregation + residual + ELU + head mean.
// TDM double-buffers the adj/adj_seq tiles into LDS (overlapped with the
// 4-head coef+WMMA work); all 4 heads' coef tiles are built in LDS with a
// single barrier, then 8 WMMAs issue back-to-back (2 barriers per m-tile).
// ---------------------------------------------------------------------------
__global__ __launch_bounds__(256) void k_aggr(const float* __restrict__ inputs,
                                              const float* __restrict__ adj,
                                              const float* __restrict__ adj_seq,
                                              const unsigned short* __restrict__ ftsT,
                                              const float* __restrict__ f1g,
                                              const float* __restrict__ f2g,
                                              const float* __restrict__ rmax,
                                              const float* __restrict__ rsum,
                                              const float* __restrict__ rmax2,
                                              const float* __restrict__ rsum2,
                                              const float* __restrict__ alpha,
                                              const float* __restrict__ bias,
                                              float* __restrict__ out) {
  const int nt = blockIdx.x % (NN / 16);
  const int b  = blockIdx.x / (NN / 16);
  const int tid  = threadIdx.x;
  const int wave = tid >> 5, lane = tid & 31, lo = lane & 15, hi = lane >> 4;
  const int rowbase = nt * 16;

  __shared__ float sh_f1[HH][16], sh_rm[HH][16], sh_ri[HH][16];
  __shared__ float sh_rm2[16], sh_ri2[16], sh_al[HH];
  __shared__ __align__(16) unsigned short s_coef[HH][16][32];
#ifdef HAVE_TDM
  __shared__ __align__(16) float s_abuf[2][16][32];   // adj tiles (double buffer)
  __shared__ __align__(16) float s_sbuf[2][16][32];   // adj_seq tiles
#endif

  if (tid < 64) {
    int h = tid >> 4, r = tid & 15;
    size_t ix = ((size_t)h * BB + b) * NN + rowbase + r;
    sh_f1[h][r] = f1g[ix];
    sh_rm[h][r] = rmax[ix];
    sh_ri[h][r] = 1.f / rsum[ix];
  }
  if (tid < 16) {
    size_t ix = (size_t)b * NN + rowbase + tid;
    sh_rm2[tid] = rmax2[ix];
    sh_ri2[tid] = 1.f / rsum2[ix];
  }
  if (tid < HH) sh_al[tid] = alpha[tid];

  v8f acc[HH][2] = {};

  // coef tile entry mapping: this thread owns (r2, k2) and (r2, k2+1)
  const int r2 = tid >> 4;
  const int k2 = (tid & 15) * 2;

  const size_t tilebase = ((size_t)b * NN + rowbase) * NN;

#ifdef HAVE_TDM
  int cur = 0;
  if (wave == 0) {   // preload tile 0
    tdm_load_tile_f32(adj     + tilebase, &s_abuf[0][0][0], 16, 32, NN, NN);
    tdm_load_tile_f32(adj_seq + tilebase, &s_sbuf[0][0][0], 16, 32, NN, NN);
    __builtin_amdgcn_s_wait_tensorcnt(0);
  }
#endif
  __syncthreads();

  for (int mb = 0; mb < NN; mb += 32) {
#ifdef HAVE_TDM
    if (wave == 0 && mb + 32 < NN) {   // async prefetch of next tile via TDM
      tdm_load_tile_f32(adj     + tilebase + mb + 32, &s_abuf[cur ^ 1][0][0], 16, 32, NN, NN);
      tdm_load_tile_f32(adj_seq + tilebase + mb + 32, &s_sbuf[cur ^ 1][0][0], 16, 32, NN, NN);
    }
    F2 av = *(const F2*)&s_abuf[cur][r2][k2];
    F2 as = *(const F2*)&s_sbuf[cur][r2][k2];
#else
    const int m0 = mb + k2;
    size_t gix = tilebase + (size_t)r2 * NN + m0;
    F2 av = *(const F2*)(adj + gix);
    F2 as = *(const F2*)(adj_seq + gix);
    if (mb + 32 < NN) {                 // global_prefetch_b8 of next tile
      __builtin_prefetch(adj + gix + 32, 0, 1);
      __builtin_prefetch(adj_seq + gix + 32, 0, 1);
    }
#endif
    // shared (adj_seq) softmax contribution
    float rm2 = sh_rm2[r2], ri2 = sh_ri2[r2];
    float sa0 = __expf(as.x - rm2) * ri2;
    float sa1 = __expf(as.y - rm2) * ri2;

    // build all 4 heads' coefficient tiles, then one barrier
#pragma unroll
    for (int h = 0; h < HH; ++h) {
      F2 f2v = *(const F2*)(f2g + ((size_t)h * BB + b) * NN + mb + k2);
      float f1v = sh_f1[h][r2], rm = sh_rm[h][r2], ri = sh_ri[h][r2], al = sh_al[h];

      float x0 = f1v + f2v.x;
      float l0 = ((x0 > 0.f) ? x0 : NEG_SLOPE * x0) + (-1e9f * (1.f - av.x));
      float c0 = al * (__expf(l0 - rm) * ri) + (1.f - al) * sa0;

      float x1 = f1v + f2v.y;
      float l1 = ((x1 > 0.f) ? x1 : NEG_SLOPE * x1) + (-1e9f * (1.f - av.y));
      float c1 = al * (__expf(l1 - rm) * ri) + (1.f - al) * sa1;

      s_coef[h][r2][k2]     = f2bf(c0);
      s_coef[h][r2][k2 + 1] = f2bf(c1);
    }
    __syncthreads();

    // 8 WMMAs back-to-back (4 heads x 2 column tiles)
#pragma unroll
    for (int h = 0; h < HH; ++h) {
      V16 A;
      const unsigned short* ap = &s_coef[h][lo][hi * 8];
      A.q[0] = *(const U4*)(ap);
      A.q[1] = *(const U4*)(ap + 16);
#pragma unroll
      for (int t = 0; t < 2; ++t) {
        const int col = wave * 32 + t * 16 + lo;
        const unsigned short* bp =
            ftsT + (((size_t)h * BB + b) * OUTC + col) * NN + mb + hi * 8;
        V16 Bv;
        Bv.q[0] = *(const U4*)(bp);
        Bv.q[1] = *(const U4*)(bp + 16);
        acc[h][t] = __builtin_amdgcn_wmma_f32_16x16x32_bf16(
            false, A.b, false, Bv.b, (short)0, acc[h][t], false, false);
      }
    }

#ifdef HAVE_TDM
    if (wave == 0) __builtin_amdgcn_s_wait_tensorcnt(0);  // next buffer ready
#endif
    __syncthreads();   // publish next tile / protect s_coef reuse
#ifdef HAVE_TDM
    cur ^= 1;
#endif
  }

  // Epilogue: vals = acc + bias[h]; out = mean_h elu(vals + inputs)
#pragma unroll
  for (int t = 0; t < 2; ++t) {
    const int col = wave * 32 + t * 16 + lo;
    float bh[HH];
#pragma unroll
    for (int h = 0; h < HH; ++h) bh[h] = bias[h * OUTC + col];
#pragma unroll
    for (int r = 0; r < 8; ++r) {
      int row = rowbase + hi * 8 + r;
      size_t oix = ((size_t)b * NN + row) * OUTC + col;
      float inv = inputs[oix];
      float s = 0.f;
#pragma unroll
      for (int h = 0; h < HH; ++h) {
        float v = acc[h][t][r] + bh[h] + inv;
        s += (v > 0.f) ? v : (__expf(v) - 1.f);
      }
      out[oix] = 0.25f * s;
    }
  }
}

// ---------------------------------------------------------------------------
// Launcher
// ---------------------------------------------------------------------------
extern "C" void kernel_launch(void* const* d_in, const int* in_sizes, int n_in,
                              void* d_out, int out_size, void* d_ws, size_t ws_size,
                              hipStream_t stream) {
  (void)in_sizes; (void)n_in; (void)out_size; (void)ws_size;

  const float* inputs  = (const float*)d_in[0];
  const float* adj     = (const float*)d_in[1];
  const float* adj_seq = (const float*)d_in[2];
  const float* W       = (const float*)d_in[3];
  const float* w1      = (const float*)d_in[4];
  const float* b1      = (const float*)d_in[5];
  const float* w2      = (const float*)d_in[6];
  const float* b2      = (const float*)d_in[7];
  const float* alpha   = (const float*)d_in[8];
  const float* bias    = (const float*)d_in[9];
  float* out = (float*)d_out;

  // Workspace layout (~10.8 MB total)
  char* p = (char*)d_ws;
  unsigned short* in_bf = (unsigned short*)p; p += (size_t)BB * NN * FF * 2;
  unsigned short* wT    = (unsigned short*)p; p += (size_t)HH * OUTC * FF * 2;
  unsigned short* ftsT  = (unsigned short*)p; p += (size_t)HH * BB * OUTC * NN * 2;
  float* f1    = (float*)p; p += (size_t)HH * BB * NN * 4;
  float* f2    = (float*)p; p += (size_t)HH * BB * NN * 4;
  float* rmaxp = (float*)p; p += (size_t)HH * BB * NN * 4;
  float* rsump = (float*)p; p += (size_t)HH * BB * NN * 4;
  float* rmax2p = (float*)p; p += (size_t)BB * NN * 4;
  float* rsum2p = (float*)p; p += (size_t)BB * NN * 4;

  k_cvt_inputs<<<(BB * NN * FF) / 256, 256, 0, stream>>>(inputs, in_bf, BB * NN * FF);
  k_tr_W<<<(HH * FF * OUTC) / 256, 256, 0, stream>>>(W, wT);
  k_fts<<<HH * BB * (NN / 16), 256, 0, stream>>>(in_bf, wT, ftsT);
  k_f1f2<<<(HH * BB * NN) / 256, 256, 0, stream>>>(ftsT, w1, b1, w2, b2, f1, f2);
  k_stats<<<BB * NN, 256, 0, stream>>>(adj, adj_seq, f1, f2, rmaxp, rsump, rmax2p, rsum2p);
  k_aggr<<<BB * (NN / 16), 256, 0, stream>>>(inputs, adj, adj_seq, ftsT, f1, f2,
                                             rmaxp, rsump, rmax2p, rsum2p,
                                             alpha, bias, out);
}
